// QuantLinear_498216206623
// MI455X (gfx1250) — compile-verified, matrix operand order checked by reading, not run on hardware
//
#include <hip/hip_runtime.h>
#include <stdint.h>

typedef __attribute__((ext_vector_type(16))) __bf16 v16bf;
typedef __attribute__((ext_vector_type(8)))  float  v8f;

#define M_DIM 8192
#define N_DIM 4096
#define K_DIM 4096
#define QGROUP 128

#define BM 256
#define BN 128
#define BK 32
#define LDSS 40   // padded LDS row stride in ushorts (80B: 16B-aligned, bank-spread)

__device__ __forceinline__ unsigned short f32_to_bf16_rne(float f) {
    union { float f; unsigned int u; } c; c.f = f;
    unsigned int u = c.u;
    unsigned int r = u + 0x7FFFu + ((u >> 16) & 1u);
    return (unsigned short)(r >> 16);
}

__device__ __forceinline__ unsigned int pack_bf16x2(float lo, float hi) {
    return (unsigned int)f32_to_bf16_rne(lo) |
           ((unsigned int)f32_to_bf16_rne(hi) << 16);
}

// LDS byte offset of a __shared__ pointer (generic addr low 32 bits == LDS offset)
__device__ __forceinline__ unsigned int lds_off(const void* p) {
    return (unsigned int)(uintptr_t)p;
}

// async copy of 16B/lane: global (SGPR base + 32-bit VGPR byte offset) -> LDS
__device__ __forceinline__ void async_copy_b128(unsigned int lds_addr,
                                                unsigned int goff,
                                                const void* sbase) {
    asm volatile("global_load_async_to_lds_b128 %0, %1, %2"
                 :
                 : "v"(lds_addr), "v"(goff), "s"(sbase)
                 : "memory");
}

// ---------------------------------------------------------------------------
// Pass 1a: group-wise (128 along K) FP4 E2M1 fake-quantize of weight -> bf16.
// One wave32 per group; mirrors reference (bucket-clipped mantissa, subnormals,
// rintf ties-to-even, scale = max|w|/6 clamped at 1e-8).
// ---------------------------------------------------------------------------
__global__ __launch_bounds__(256)
void dequant_fp4_kernel(const float* __restrict__ w,
                        unsigned short* __restrict__ wq) {
    const int wid  = threadIdx.x >> 5;
    const int lane = threadIdx.x & 31;
    const long group = (long)blockIdx.x * 8 + wid;
    const int row = (int)(group >> 5);
    const int g   = (int)(group & 31);
    const long base = (long)row * K_DIM + (long)g * QGROUP + lane * 4;

    const float4 v = *(const float4*)(w + base);
    float amax = fmaxf(fmaxf(fabsf(v.x), fabsf(v.y)),
                       fmaxf(fabsf(v.z), fabsf(v.w)));
    #pragma unroll
    for (int m = 16; m >= 1; m >>= 1)
        amax = fmaxf(amax, __shfl_xor(amax, m, 32));

    const float scale = fmaxf(amax * (1.0f / 6.0f), 1e-8f);
    const float inv   = 1.0f / scale;

    float e4[4] = {v.x, v.y, v.z, v.w};
    unsigned short q[4];
    #pragma unroll
    for (int i = 0; i < 4; ++i) {
        const float xv = e4[i] * inv;
        const float xa = fabsf(xv);
        const bool  isz = (xa == 0.0f);
        const float xs  = isz ? 1e-8f : xa;
        int e = (int)floorf(log2f(xs));
        const bool sub = (e < 0);
        int ec = e < 0 ? 0 : (e > 2 ? 2 : e);
        const float p2 = exp2f((float)ec);
        float mn = rintf((xs / p2 - 1.0f) * 2.0f);
        mn = fminf(fmaxf(mn, 0.0f), 1.0f);
        float ms = rintf(xs * 2.0f);
        ms = fminf(fmaxf(ms, 0.0f), 1.0f);
        const float mant = sub ? ms : mn;
        float mag = sub ? (mant * 0.5f) : ((1.0f + mant * 0.5f) * p2);
        float dq  = isz ? 0.0f : (xv < 0.0f ? -mag : mag);
        q[i] = f32_to_bf16_rne(dq * scale);
    }
    ushort4 packed;
    packed.x = q[0]; packed.y = q[1]; packed.z = q[2]; packed.w = q[3];
    *(ushort4*)(wq + base) = packed;
}

// ---------------------------------------------------------------------------
// Pass 1b: x fp32 -> bf16 (one-time, bandwidth-bound; hoists conversion out of
// the GEMM hot loop which otherwise redoes it once per N-block = 32x).
// ---------------------------------------------------------------------------
__global__ __launch_bounds__(256)
void cvt_x_bf16_kernel(const float* __restrict__ x,
                       unsigned short* __restrict__ xq) {
    const long i = ((long)blockIdx.x * 256 + threadIdx.x) * 8;
    const float4 a0 = *(const float4*)(x + i);
    const float4 a1 = *(const float4*)(x + i + 4);
    uint4 p;
    p.x = pack_bf16x2(a0.x, a0.y);
    p.y = pack_bf16x2(a0.z, a0.w);
    p.z = pack_bf16x2(a1.x, a1.y);
    p.w = pack_bf16x2(a1.z, a1.w);
    *(uint4*)(xq + i) = p;
}

// ---------------------------------------------------------------------------
// Pass 2: bf16 WMMA GEMM, async-to-LDS double buffered, 4x4 wave tiles.
// out[m,n] = sum_k xq[m,k]*wq[n,k] + bias[n]
// Block: 256x128, BK=32, 256 threads = 8 waves as 4(M) x 2(N);
// each wave owns a 64x64 patch = 4x4 16x16 f32 accumulators -> 16 WMMA/k-step
// against 16 ds_load_b128 (1.0 LDS loads per WMMA vs 1.5 before).
// Per k-step per wave: 6 async b128 (prefetch k+1), s_wait_asynccnt, barrier,
// 16 ds_load_b128, 16 WMMA, barrier.
// ---------------------------------------------------------------------------
__global__ __launch_bounds__(256)
void gemm_bf16_wmma(const unsigned short* __restrict__ xq,
                    const unsigned short* __restrict__ wq,
                    const float* __restrict__ bias,
                    float* __restrict__ out) {
    __shared__ unsigned short sA[2][BM * LDSS];   // 2 x 20 KB
    __shared__ unsigned short sB[2][BN * LDSS];   // 2 x 10 KB

    const int tid    = threadIdx.x;
    const int lane   = tid & 31;
    const int wid    = tid >> 5;
    const int wave_m = wid & 3;   // 0..3 -> 64 rows each
    const int wave_n = wid >> 2;  // 0..1 -> 64 cols each

    const long block_m = (long)blockIdx.y * BM;
    const long block_n = (long)blockIdx.x * BN;

    // staging: 16B chunks; A tile = 1024 chunks (4/thread), B tile = 512 (2/thread)
    unsigned int gA[4], lA[4], gB[2], lB[2];
    #pragma unroll
    for (int i = 0; i < 4; ++i) {
        const int c = tid + i * 256;
        const int row = c >> 2, cs = (c & 3) * 8;
        gA[i] = (unsigned int)(((block_m + row) * (long)K_DIM + cs) * 2);
        lA[i] = (unsigned int)((row * LDSS + cs) * 2);
    }
    #pragma unroll
    for (int i = 0; i < 2; ++i) {
        const int c = tid + i * 256;
        const int row = c >> 2, cs = (c & 3) * 8;
        gB[i] = (unsigned int)(((block_n + row) * (long)K_DIM + cs) * 2);
        lB[i] = (unsigned int)((row * LDSS + cs) * 2);
    }

    v8f acc[4][4];
    #pragma unroll
    for (int tm = 0; tm < 4; ++tm)
        #pragma unroll
        for (int tn = 0; tn < 4; ++tn)
            acc[tm][tn] = (v8f){0.f, 0.f, 0.f, 0.f, 0.f, 0.f, 0.f, 0.f};

    // fragment addressing (ISA 16-bit A/B layout): lane L -> row L%16,
    // lanes 0-15: K=[0..7]+[16..23]; lanes 16-31: K=[8..15]+[24..31]
    const int fr  = lane & 15;
    const int fkh = (lane >> 4) << 3;

    union Frag { uint4 u[2]; v16bf v; };

    auto issue_tile = [&](int k0, int buf) {
        const unsigned int kb = (unsigned int)(k0 * 2);
        const unsigned int a0 = lds_off(&sA[buf][0]);
        const unsigned int b0 = lds_off(&sB[buf][0]);
        #pragma unroll
        for (int i = 0; i < 4; ++i)
            async_copy_b128(a0 + lA[i], gA[i] + kb, xq);
        #pragma unroll
        for (int i = 0; i < 2; ++i)
            async_copy_b128(b0 + lB[i], gB[i] + kb, wq);
    };

    int parity = 0;
    issue_tile(0, 0);

    for (int k0 = 0; k0 < K_DIM; k0 += BK) {
        if (k0 + BK < K_DIM) {
            issue_tile(k0 + BK, parity ^ 1);                    // prefetch next
            asm volatile("s_wait_asynccnt 0x6" ::: "memory");   // current done
        } else {
            asm volatile("s_wait_asynccnt 0x0" ::: "memory");
        }
        __syncthreads();

        Frag fb[4];
        #pragma unroll
        for (int tn = 0; tn < 4; ++tn) {
            const int row = wave_n * 64 + tn * 16 + fr;
            const unsigned short* p = &sB[parity][row * LDSS + fkh];
            fb[tn].u[0] = *(const uint4*)(p);
            fb[tn].u[1] = *(const uint4*)(p + 16);
        }
        #pragma unroll
        for (int tm = 0; tm < 4; ++tm) {
            Frag fa;
            const int row = wave_m * 64 + tm * 16 + fr;
            const unsigned short* p = &sA[parity][row * LDSS + fkh];
            fa.u[0] = *(const uint4*)(p);
            fa.u[1] = *(const uint4*)(p + 16);
            #pragma unroll
            for (int tn = 0; tn < 4; ++tn)
                acc[tm][tn] = __builtin_amdgcn_wmma_f32_16x16x32_bf16(
                    false, fa.v, false, fb[tn].v,
                    (short)0, acc[tm][tn], false, false);
        }

        __syncthreads();   // all waves done reading buf[parity] -> reusable
        parity ^= 1;
    }

    // epilogue: bias add + store (C layout: VGPR g -> M=+g / +8+g, N=lane%16)
    #pragma unroll
    for (int tn = 0; tn < 4; ++tn) {
        const long gn = block_n + wave_n * 64 + tn * 16 + (lane & 15);
        const float bv = bias[gn];
        #pragma unroll
        for (int tm = 0; tm < 4; ++tm) {
            const long gm = block_m + wave_m * 64 + tm * 16 + ((lane >> 4) << 3);
            #pragma unroll
            for (int g = 0; g < 8; ++g)
                out[(gm + g) * (long)N_DIM + gn] = acc[tm][tn][g] + bv;
        }
    }
}

extern "C" void kernel_launch(void* const* d_in, const int* in_sizes, int n_in,
                              void* d_out, int out_size, void* d_ws, size_t ws_size,
                              hipStream_t stream) {
    const float* x    = (const float*)d_in[0];   // (4,2048,4096) f32
    const float* w    = (const float*)d_in[1];   // (4096,4096)  f32
    const float* bias = (const float*)d_in[2];   // (4096,)      f32
    float* out        = (float*)d_out;           // (4,2048,4096) f32

    unsigned short* wq = (unsigned short*)d_ws;                    // 32 MB bf16 weight
    unsigned short* xq = wq + (size_t)N_DIM * K_DIM;               // 64 MB bf16 x

    dequant_fp4_kernel<<<(N_DIM * (K_DIM / QGROUP)) / 8, 256, 0, stream>>>(w, wq);
    cvt_x_bf16_kernel<<<(int)(((long)M_DIM * K_DIM / 8) / 256), 256, 0, stream>>>(x, xq);

    dim3 grid(N_DIM / BN, M_DIM / BM);
    gemm_bf16_wmma<<<grid, 256, 0, stream>>>(xq, wq, bias, out);
}
